// NTXentLoss_50972671869123
// MI455X (gfx1250) — compile-verified, matrix operand order checked by reading, not run on hardware
//
#include <hip/hip_runtime.h>
#include <math.h>

// NT-Xent loss with hard-negative reweighting, B=8192, D=512, T=0.1, w=0.5.
// 3 WMMA GEMM passes (f16 in, f32 accum) over recomputed sim tiles:
//   Pass 1: per-row argmax of sim excluding diagonal      (atomicMax u64 key)
//   Pass 2: per-row / per-col max of scaled logits        (atomicMax u32 key)
//   Pass 3: per-row / per-col sum exp(l - max)            (deterministic partials)
//   Final : loss = sum_i(rowLSE_i + colLSE_i - 2*diag_i) / (2B)
// Inputs pre-converted to f16 once; GEMM stages K-chunks into double-buffered
// LDS via gfx1250 async global->LDS loads (ASYNCcnt), WMMAs overlap the DMA.
// All reductions are order-deterministic.

#define B_SZ 8192
#define D_SZ 512
#define INV_T 10.0f
#define HNW 0.5f

typedef __attribute__((ext_vector_type(16))) _Float16 v16h;
typedef __attribute__((ext_vector_type(8)))  _Float16 v8h;
typedef __attribute__((ext_vector_type(8)))  float    v8f;

// ---- monotonic float <-> unsigned key (order-preserving) ----
__device__ __forceinline__ unsigned key32(float f) {
  unsigned u = __float_as_uint(f);
  return (u & 0x80000000u) ? ~u : (u | 0x80000000u);
}
__device__ __forceinline__ float unkey32(unsigned k) {
  unsigned u = (k & 0x80000000u) ? (k & 0x7fffffffu) : ~k;
  return __uint_as_float(u);
}

// ---- load one 16x32 f16 A/B fragment from LDS (row = lane&15, K per ISA layout)
// lanes 0-15: K 0..7 then 16..23 ; lanes 16-31: K 8..15 then 24..31
__device__ __forceinline__ v16h load_frag(const _Float16* row, int lane) {
  const int hsel = (lane >> 4) & 1;
  union { v16h h; float4 f[2]; } u;
  u.f[0] = *(const float4*)(row + hsel * 8);
  u.f[1] = *(const float4*)(row + hsel * 8 + 16);
  return u.h;
}

// ---- async global -> LDS stage of 64 contiguous bytes (4x b128).
// Per CDNA5 ISA, the instruction offset is added to BOTH the LDS address
// (VDST vgpr) and the global address, so one vdst/vaddr pair covers 64B.
__device__ __forceinline__ void async_stage64(unsigned ldsoff, const _Float16* g) {
  asm volatile(
      "global_load_async_to_lds_b128 %0, %1, off\n\t"
      "global_load_async_to_lds_b128 %0, %1, off offset:16\n\t"
      "global_load_async_to_lds_b128 %0, %1, off offset:32\n\t"
      "global_load_async_to_lds_b128 %0, %1, off offset:48"
      :: "v"(ldsoff), "v"(g) : "memory");
}
__device__ __forceinline__ void wait_async0() {
  asm volatile("s_wait_asynccnt 0x0" ::: "memory");
}

// Single shared block => LDS segment offset 0; static byte offsets below.
struct __align__(16) Smem {
  _Float16 A[2][128][40];   // byte off: p*10240 + row*80        (20480 B)
  _Float16 B[2][128][40];   // byte off: 20480 + p*10240 + row*80 (20480 B)
  float    row[128][2];
  float    col[128][2];
};
#define LDS_A_OFF(p, r) ((unsigned)((p) * 10240u + (r) * 80u))
#define LDS_B_OFF(p, r) ((unsigned)(20480u + (p) * 10240u + (r) * 80u))

// =====================================================================
// GEMM pass: 128x128 block tile, 4 waves each owning 64x64 (4x4 WMMA frags),
// K staged via double-buffered LDS with async DMA, single barrier per k-step.
// MODE 1: row argmax (excl diag) ; MODE 2: scaled row/col max ;
// MODE 3: scaled row/col exp-sums into deterministic partial arrays.
// =====================================================================
template<int MODE>
__global__ __launch_bounds__(128)
void gemm_pass(const _Float16* __restrict__ Ih, const _Float16* __restrict__ Sh,
               unsigned long long* __restrict__ rowKey,
               unsigned* __restrict__ rowMaxKey, unsigned* __restrict__ colMaxKey,
               float* __restrict__ rowPartial, float* __restrict__ colPartial)
{
  __shared__ Smem sm;

  const int tid   = threadIdx.x;
  const int lane  = tid & 31;
  const int w     = tid >> 5;          // 0..3
  const int waveM = w >> 1;            // 0..1  (64 rows each)
  const int waveN = w & 1;             // 0..1  (64 cols each)
  const int blockM = blockIdx.y, blockN = blockIdx.x;
  const int rowBlock = blockM * 128, colBlock = blockN * 128;

  v8f zero = {};
  v8f acc[4][4];
  #pragma unroll
  for (int mi = 0; mi < 4; mi++)
    #pragma unroll
    for (int ni = 0; ni < 4; ni++) acc[mi][ni] = zero;

  // staging: thread t stages row t of both tiles (32 halves = 64B)
  const _Float16* gA = Ih + (size_t)(rowBlock + tid) * D_SZ;
  const _Float16* gB = Sh + (size_t)(colBlock + tid) * D_SZ;

  // preload chunk 0 into buffer 0 via async DMA
  async_stage64(LDS_A_OFF(0, tid), gA);
  async_stage64(LDS_B_OFF(0, tid), gB);
  wait_async0();
  __syncthreads();

  const int nk = D_SZ / 32;            // 16 k-steps
  for (int ks = 0; ks < nk; ks++) {
    const int p = ks & 1;

    // kick off next chunk's async DMA into the other buffer (hides under WMMA)
    if (ks + 1 < nk) {
      async_stage64(LDS_A_OFF(p ^ 1, tid), gA + (ks + 1) * 32);
      async_stage64(LDS_B_OFF(p ^ 1, tid), gB + (ks + 1) * 32);
      if (ks + 2 < nk) {
        __builtin_prefetch(gA + (ks + 2) * 32, 0, 0);  // global_prefetch_b8
        __builtin_prefetch(gB + (ks + 2) * 32, 0, 0);
      }
    }

    v16h afr[4], bfr[4];
    #pragma unroll
    for (int mi = 0; mi < 4; mi++)
      afr[mi] = load_frag(&sm.A[p][waveM * 64 + mi * 16 + (lane & 15)][0], lane);
    #pragma unroll
    for (int ni = 0; ni < 4; ni++)
      bfr[ni] = load_frag(&sm.B[p][waveN * 64 + ni * 16 + (lane & 15)][0], lane);

    #pragma unroll
    for (int mi = 0; mi < 4; mi++)
      #pragma unroll
      for (int ni = 0; ni < 4; ni++)
        acc[mi][ni] = __builtin_amdgcn_wmma_f32_16x16x32_f16(
            false, afr[mi], false, bfr[ni], (short)0, acc[mi][ni], false, false);

    if (ks + 1 < nk) wait_async0();
    __syncthreads();
  }

  // ---- epilogue ----
  const int hsel = lane >> 4;   // C frag: lanes>=16 hold rows M+8
  const int n0   = lane & 15;   // C frag: lane holds column N = lane&15

  if (MODE == 1) {
    #pragma unroll
    for (int mi = 0; mi < 4; mi++)
      #pragma unroll
      for (int v = 0; v < 8; v++) {
        const int rowG = rowBlock + waveM * 64 + mi * 16 + hsel * 8 + v;
        float best = -__builtin_inff(); int bidx = 0;
        #pragma unroll
        for (int ni = 0; ni < 4; ni++) {
          const int colG = colBlock + waveN * 64 + ni * 16 + n0;
          const float x = acc[mi][ni][v] * INV_T;
          if (colG != rowG && x > best) { best = x; bidx = colG; }
        }
        #pragma unroll
        for (int off = 1; off < 16; off <<= 1) {
          const float ov = __shfl_xor(best, off, 32);
          const int   oi = __shfl_xor(bidx, off, 32);
          if (ov > best || (ov == best && oi > bidx)) { best = ov; bidx = oi; }
        }
        if (n0 == 0) {
          const unsigned long long k =
              ((unsigned long long)key32(best) << 32) | (unsigned)bidx;
          atomicMax(&rowKey[rowG], k);
        }
      }
    return;
  }

  // MODE >= 2: apply 1/T and hard-negative 1.5x on column == hardest(row)
  #pragma unroll
  for (int mi = 0; mi < 4; mi++)
    #pragma unroll
    for (int v = 0; v < 8; v++) {
      const int rowG = rowBlock + waveM * 64 + mi * 16 + hsel * 8 + v;
      const int hrd  = (int)(rowKey[rowG] & 0xffffffffull);
      #pragma unroll
      for (int ni = 0; ni < 4; ni++) {
        const int colG = colBlock + waveN * 64 + ni * 16 + n0;
        float x = acc[mi][ni][v] * INV_T;
        if (colG == hrd) x *= (1.0f + HNW);
        acc[mi][ni][v] = x;
      }
    }

  if (MODE == 2) {
    // row max of scaled logits (includes diagonal)
    #pragma unroll
    for (int mi = 0; mi < 4; mi++)
      #pragma unroll
      for (int v = 0; v < 8; v++) {
        const int rowG = rowBlock + waveM * 64 + mi * 16 + hsel * 8 + v;
        float m = -__builtin_inff();
        #pragma unroll
        for (int ni = 0; ni < 4; ni++) m = fmaxf(m, acc[mi][ni][v]);
        #pragma unroll
        for (int off = 1; off < 16; off <<= 1) m = fmaxf(m, __shfl_xor(m, off, 32));
        if (n0 == 0) atomicMax(&rowMaxKey[rowG], key32(m));
      }
    // col max of scaled logits
    #pragma unroll
    for (int ni = 0; ni < 4; ni++) {
      const int colG = colBlock + waveN * 64 + ni * 16 + n0;
      float m = -__builtin_inff();
      #pragma unroll
      for (int mi = 0; mi < 4; mi++)
        #pragma unroll
        for (int v = 0; v < 8; v++) m = fmaxf(m, acc[mi][ni][v]);
      m = fmaxf(m, __shfl_xor(m, 16, 32));
      if (hsel == 0) atomicMax(&colMaxKey[colG], key32(m));
    }
    return;
  }

  if (MODE == 3) {
    // row exp-sums
    #pragma unroll
    for (int mi = 0; mi < 4; mi++)
      #pragma unroll
      for (int v = 0; v < 8; v++) {
        const int rowG = rowBlock + waveM * 64 + mi * 16 + hsel * 8 + v;
        const float rm = unkey32(rowMaxKey[rowG]);
        float s = 0.0f;
        #pragma unroll
        for (int ni = 0; ni < 4; ni++) s += __expf(acc[mi][ni][v] - rm);
        #pragma unroll
        for (int off = 1; off < 16; off <<= 1) s += __shfl_xor(s, off, 32);
        if (n0 == 0) sm.row[waveM * 64 + mi * 16 + hsel * 8 + v][waveN] = s;
      }
    // col exp-sums
    #pragma unroll
    for (int ni = 0; ni < 4; ni++) {
      const int colG = colBlock + waveN * 64 + ni * 16 + n0;
      const float cm = unkey32(colMaxKey[colG]);
      float s = 0.0f;
      #pragma unroll
      for (int mi = 0; mi < 4; mi++)
        #pragma unroll
        for (int v = 0; v < 8; v++) s += __expf(acc[mi][ni][v] - cm);
      s += __shfl_xor(s, 16, 32);
      if (hsel == 0) sm.col[waveN * 64 + ni * 16 + n0][waveM] = s;
    }
    __syncthreads();
    {
      const int rowG = rowBlock + tid;
      rowPartial[(size_t)rowG * 64 + blockN] = sm.row[tid][0] + sm.row[tid][1];
      const int colG = colBlock + tid;
      colPartial[(size_t)colG * 64 + blockM] = sm.col[tid][0] + sm.col[tid][1];
    }
  }
}

// =====================================================================
// One-shot f32 -> f16 conversion of both embedding matrices (8 elems/thread).
__global__ __launch_bounds__(256)
void cvt_f16_kernel(const float* __restrict__ I, const float* __restrict__ S,
                    _Float16* __restrict__ Ih, _Float16* __restrict__ Sh) {
  const size_t i = ((size_t)blockIdx.x * 256 + threadIdx.x) * 8;
  {
    float4 f0 = ((const float4*)(I + i))[0];
    float4 f1 = ((const float4*)(I + i))[1];
    v8h h = {(_Float16)f0.x,(_Float16)f0.y,(_Float16)f0.z,(_Float16)f0.w,
             (_Float16)f1.x,(_Float16)f1.y,(_Float16)f1.z,(_Float16)f1.w};
    union { v8h h; float4 f; } u; u.h = h;
    *(float4*)(Ih + i) = u.f;
  }
  {
    float4 f0 = ((const float4*)(S + i))[0];
    float4 f1 = ((const float4*)(S + i))[1];
    v8h h = {(_Float16)f0.x,(_Float16)f0.y,(_Float16)f0.z,(_Float16)f0.w,
             (_Float16)f1.x,(_Float16)f1.y,(_Float16)f1.z,(_Float16)f1.w};
    union { v8h h; float4 f; } u; u.h = h;
    *(float4*)(Sh + i) = u.f;
  }
}

__global__ void init_kernel(unsigned long long* rowKey,
                            unsigned* rowMaxKey, unsigned* colMaxKey) {
  const int i = blockIdx.x * 256 + threadIdx.x;
  if (i < B_SZ) { rowKey[i] = 0ull; rowMaxKey[i] = 0u; colMaxKey[i] = 0u; }
}

// diag[i] = dot(I_i, S_i) / T  (full f32 precision, one wave per row)
__global__ __launch_bounds__(256) void diag_kernel(const float* __restrict__ I,
                                                   const float* __restrict__ S,
                                                   float* __restrict__ diag) {
  const int w = threadIdx.x >> 5, lane = threadIdx.x & 31;
  const int row = blockIdx.x * 8 + w;
  const float* a = I + (size_t)row * D_SZ;
  const float* b = S + (size_t)row * D_SZ;
  float s = 0.0f;
  for (int j = lane; j < D_SZ; j += 32) s += a[j] * b[j];
  #pragma unroll
  for (int off = 1; off < 32; off <<= 1) s += __shfl_xor(s, off, 32);
  if (lane == 0) diag[row] = s * INV_T;
}

__global__ __launch_bounds__(256)
void final_kernel(const unsigned* __restrict__ rowMaxKey,
                  const unsigned* __restrict__ colMaxKey,
                  const float* __restrict__ rowPartial,
                  const float* __restrict__ colPartial,
                  const float* __restrict__ diag, float* __restrict__ out) {
  __shared__ double red[256];
  double acc = 0.0;
  for (int i = threadIdx.x; i < B_SZ; i += 256) {
    float rs = 0.0f, cs = 0.0f;
    for (int t = 0; t < 64; t++) rs += rowPartial[(size_t)i * 64 + t];
    for (int t = 0; t < 64; t++) cs += colPartial[(size_t)i * 64 + t];
    const float term = unkey32(rowMaxKey[i]) + logf(rs) +
                       unkey32(colMaxKey[i]) + logf(cs) - 2.0f * diag[i];
    acc += (double)term;
  }
  red[threadIdx.x] = acc;
  __syncthreads();
  for (int s = 128; s > 0; s >>= 1) {
    if (threadIdx.x < s) red[threadIdx.x] += red[threadIdx.x + s];
    __syncthreads();
  }
  if (threadIdx.x == 0) out[0] = (float)(red[0] / (2.0 * (double)B_SZ));
}

// =====================================================================
extern "C" void kernel_launch(void* const* d_in, const int* in_sizes, int n_in,
                              void* d_out, int out_size, void* d_ws, size_t ws_size,
                              hipStream_t stream) {
  (void)in_sizes; (void)n_in; (void)out_size; (void)ws_size;
  const float* I = (const float*)d_in[0];   // image_embeddings [8192,512] f32
  const float* S = (const float*)d_in[1];   // song_embeddings  [8192,512] f32

  char* ws = (char*)d_ws;                   // ~20.2 MB used
  _Float16* Ih = (_Float16*)(ws);                                       // 8 MB
  _Float16* Sh = (_Float16*)(ws + 8388608);                             // 8 MB
  char* ws2 = ws + 16777216;
  unsigned long long* rowKey    = (unsigned long long*)(ws2);           // 64 KB
  unsigned*           rowMaxKey = (unsigned*)(ws2 + 65536);             // 32 KB
  unsigned*           colMaxKey = (unsigned*)(ws2 + 98304);             // 32 KB
  float*              diag      = (float*)(ws2 + 131072);               // 32 KB
  float*              rowPart   = (float*)(ws2 + 163840);               // 2 MB
  float*              colPart   = (float*)(ws2 + 163840 + 2097152);     // 2 MB

  cvt_f16_kernel<<<(B_SZ * D_SZ) / (256 * 8), 256, 0, stream>>>(I, S, Ih, Sh);
  init_kernel<<<B_SZ / 256, 256, 0, stream>>>(rowKey, rowMaxKey, colMaxKey);
  diag_kernel<<<B_SZ / 8, 256, 0, stream>>>(I, S, diag);

  dim3 grid(B_SZ / 128, B_SZ / 128);        // 64 x 64 block tiles
  gemm_pass<1><<<grid, 128, 0, stream>>>(Ih, Sh, rowKey, rowMaxKey, colMaxKey, rowPart, colPart);
  gemm_pass<2><<<grid, 128, 0, stream>>>(Ih, Sh, rowKey, rowMaxKey, colMaxKey, rowPart, colPart);
  gemm_pass<3><<<grid, 128, 0, stream>>>(Ih, Sh, rowKey, rowMaxKey, colMaxKey, rowPart, colPart);

  final_kernel<<<1, 256, 0, stream>>>(rowMaxKey, colMaxKey, rowPart, colPart, diag, (float*)d_out);
}
